// NeRank_31069793419532
// MI455X (gfx1250) — matrix operand order repaired and checked by reading, not compiled, not run on hardware
//
#include <hip/hip_runtime.h>
#include <hip/hip_bf16.h>

// ---------------------------------------------------------------------------
// NeRank forward on MI455X (gfx1250).
//
// BiLSTM recurrence is done ONE TIMESTEP PER KERNEL LAUNCH (global sync at the
// kernel boundary).  Each block computes a [64 rows x 64 hidden] slice with
// all 4 gates (i,f,g,o) via v_wmma_f32_16x16x32_f16 and fuses the LSTM
// pointwise + state update.  A block therefore reads only its 256 KB weight
// slice, cutting L2 weight traffic ~2000x vs. a fused per-block recurrence.
// All 4 independent recurrences (u/v weight sets x fwd/bwd) run in one grid.
// ---------------------------------------------------------------------------

typedef __attribute__((ext_vector_type(16))) _Float16 v16h;
typedef __attribute__((ext_vector_type(8)))  float    v8f;

__device__ __forceinline__ v8f wmma16(v16h a, v16h b, v8f c) {
  // (neg_a, A, neg_b, B, c_mod, C, reuse_a, reuse_b)
  return __builtin_amdgcn_wmma_f32_16x16x32_f16(false, a, false, b,
                                                (short)0, c, false, false);
}

// A fragment (16x32 f16, row-major source).  lane: m = lane&15, hi = lane>>4.
// Slots 0..15 = halves [0..15] of base ->
// two aligned 16B LDS loads.  caller passes base = &src[m][kb*32 + hi*16].
__device__ __forceinline__ v16h load_a_frag(const _Float16* base) {
  v16h a;
#pragma unroll
  for (int p = 0; p < 8; ++p) {
    const int k = 2 * p;   // 0,2,4,...,14
    a[2 * p]     = base[k];
    a[2 * p + 1] = base[k + 1];
  }
  return a;
}

// B fragment for B = W^T, W row-major [1024][256] f16.  B[k][n] = W[n][k].
// lane: W row nrow, same for every lane, holds 16 consecutive K -> 32B contiguous load.
__device__ __forceinline__ v16h load_b_frag(const _Float16* __restrict__ W,
                                            int nrow, int kb, int hi) {
  return *reinterpret_cast<const v16h*>(W + (size_t)nrow * 256 + kb * 32 + hi * 16);
}

__device__ __forceinline__ float sigm(float x) { return 1.0f / (1.0f + __expf(-x)); }
// branch-free tanh (saturates correctly at +-1 for large |x|)
__device__ __forceinline__ float tanh_fast(float x) {
  return 1.0f - 2.0f / (__expf(2.0f * x) + 1.0f);
}
__device__ __forceinline__ float logsig(float x) {
  return fminf(x, 0.0f) - log1pf(__expf(-fabsf(x)));
}
__device__ __forceinline__ float relu(float x) { return fmaxf(x, 0.0f); }

// ---------------------------------------------------------------------------
// fp32 -> fp16 weight conversion
// ---------------------------------------------------------------------------
__global__ void cvt_f32_f16(const float* __restrict__ src,
                            _Float16* __restrict__ dst, int n) {
  int i = blockIdx.x * blockDim.x + threadIdx.x;
  if (i < n) dst[i] = (_Float16)src[i];
}

// ---------------------------------------------------------------------------
// build combined row -> sequence index / length tables (8192 rows)
// rows 0..1023: qpos_u | 1024..2047: rank_q | 2048..3071: qpos_v | rest: qpos_neg
// ---------------------------------------------------------------------------
__global__ void build_idx(const int* __restrict__ qpos_u, const int* __restrict__ rank_q,
                          const int* __restrict__ qpos_v, const int* __restrict__ qpos_neg,
                          const int* __restrict__ q_lens,
                          int* __restrict__ seqidx, int* __restrict__ lens) {
  int i = blockIdx.x * blockDim.x + threadIdx.x;
  if (i >= 8192) return;
  int s;
  if (i < 1024)      s = qpos_u[i];
  else if (i < 2048) s = rank_q[i - 1024];
  else if (i < 3072) s = qpos_v[i - 2048];
  else               s = qpos_neg[i - 3072];
  seqidx[i] = s;
  lens[i]   = q_lens[s];
}

// zero h (2 dir x 2 pp x 8192 x 256 f16) and c (2 dir x 8192 x 256 f32)
__global__ void zero_state(_Float16* __restrict__ h, float* __restrict__ c) {
  size_t i = (size_t)blockIdx.x * blockDim.x + threadIdx.x;   // 8,388,608
  h[i] = (_Float16)0.0f;
  if (i < (size_t)2 * 8192 * 256) c[i] = 0.0f;
}

// ---------------------------------------------------------------------------
// One LSTM timestep for all 4 recurrences.
// grid = 1024 blocks x 128 threads:
//   b>>9   : dir (0 fwd, 1 bwd)
//   b&511  : r<128 -> u-set, mb=r>>2 (rows mb*64), hg=r&3
//            r>=128 -> v-set, mb=32+((r-128)>>2), hg=(r-128)&3
// Each block: rows [mb*64, +64), hidden cols [hg*64, +64), all 4 gates.
// ---------------------------------------------------------------------------
__launch_bounds__(128)
__global__ void lstm_step(const float* __restrict__ q_texts,     // [4096,64,256]
                          const int* __restrict__ seqidx,        // [8192]
                          const int* __restrict__ lens,          // [8192]
                          int step,
                          const _Float16* __restrict__ w16,      // 8 x [1024,256]
                          const float* __restrict__ ub_f, const float* __restrict__ ub_b,
                          const float* __restrict__ vb_f, const float* __restrict__ vb_b,
                          _Float16* __restrict__ hbase,          // [2dir][2pp][8192][256]
                          float* __restrict__ cbase) {           // [2dir][8192][256]
  __shared__ _Float16 xs[64][256];   // x_t tile (32 KB)
  __shared__ _Float16 hs[64][256];   // h_cur tile (32 KB)

  const int tid  = threadIdx.x;
  const int lane = tid & 31;
  const int wave = tid >> 5;
  const int hi   = lane >> 4;
  const int col  = lane & 15;

  const int b   = blockIdx.x;
  const int dir = b >> 9;
  int r = b & 511;
  int set, mb, hg;
  if (r < 128) { set = 0; mb = r >> 2;              hg = r & 3; }
  else         { set = 1; mb = 32 + ((r - 128) >> 2); hg = (r - 128) & 3; }

  const int row0 = mb * 64;
  const int tt   = dir ? (63 - step) : step;
  const int cur  = step & 1;
  const int nxt  = cur ^ 1;

  const size_t WN = (size_t)1024 * 256;
  const _Float16* Wih = w16 + ((size_t)set * 4 + (size_t)dir * 2 + 0) * WN;
  const _Float16* Whh = w16 + ((size_t)set * 4 + (size_t)dir * 2 + 1) * WN;
  const float* bias = set == 0 ? (dir ? ub_b : ub_f) : (dir ? vb_b : vb_f);

  _Float16* hcur = hbase + ((size_t)dir * 2 + cur) * (8192 * 256);
  _Float16* hnxt = hbase + ((size_t)dir * 2 + nxt) * (8192 * 256);
  float*    cst  = cbase + (size_t)dir * (8192 * 256);

  // ---- stage x_t (f32 -> f16) and h_cur into LDS ----
  for (int j = tid; j < 64 * 256; j += 128) {
    int m = j >> 8, d = j & 255;
    xs[m][d] = (_Float16)q_texts[(size_t)seqidx[row0 + m] * 16384 + (size_t)tt * 256 + d];
  }
  {
    const unsigned* src = (const unsigned*)(hcur + (size_t)row0 * 256);
    unsigned* dst = (unsigned*)&hs[0][0];
    for (int j = tid; j < 64 * 256 / 2; j += 128) dst[j] = src[j];
  }
  __syncthreads();

  // ---- gates: [64,256]x(WihT slice) + [64,256]x(WhhT slice) ----
  const int hid0 = hg * 64 + wave * 16;     // global hidden base for this wave
  v8f acc[4][4] = {};                       // [gate i/f/g/o][M-tile]

#pragma unroll 1
  for (int kb = 0; kb < 8; ++kb) {          // x @ Wih^T
    v16h a[4];
#pragma unroll
    for (int mt = 0; mt < 4; ++mt)
      a[mt] = load_a_frag(&xs[mt * 16 + col][kb * 32 + hi * 16]);
#pragma unroll
    for (int g = 0; g < 4; ++g) {
      v16h bw = load_b_frag(Wih, g * 256 + hid0, kb, hi);
#pragma unroll
      for (int mt = 0; mt < 4; ++mt) acc[g][mt] = wmma16(a[mt], bw, acc[g][mt]);
    }
  }
#pragma unroll 1
  for (int kb = 0; kb < 8; ++kb) {          // h @ Whh^T
    v16h a[4];
#pragma unroll
    for (int mt = 0; mt < 4; ++mt)
      a[mt] = load_a_frag(&hs[mt * 16 + col][kb * 32 + hi * 16]);
#pragma unroll
    for (int g = 0; g < 4; ++g) {
      v16h bw = load_b_frag(Whh, g * 256 + hid0, kb, hi);
#pragma unroll
      for (int mt = 0; mt < 4; ++mt) acc[g][mt] = wmma16(a[mt], bw, acc[g][mt]);
    }
  }

  // ---- bias + LSTM pointwise + state update ----
  const int hidg = hid0 + col;
  const float bI  = bias[hidg];
  const float bFg = bias[256 + hidg];
  const float bG  = bias[512 + hidg];
  const float bO  = bias[768 + hidg];

#pragma unroll
  for (int mt = 0; mt < 4; ++mt) {
#pragma unroll
    for (int rr = 0; rr < 8; ++rr) {
      const int m   = mt * 16 + hi * 8 + rr;     // local row for this C slot
      const int row = row0 + m;
      const size_t ci = (size_t)row * 256 + hidg;
      float si = sigm(acc[0][mt][rr] + bI);
      float sf = sigm(acc[1][mt][rr] + bFg);
      float tg = tanh_fast(acc[2][mt][rr] + bG);
      float so = sigm(acc[3][mt][rr] + bO);
      float cold = cst[ci];
      float c2 = sf * cold + si * tg;
      float h2 = so * tanh_fast(c2);
      bool  on = (tt < lens[row]);
      if (on) cst[ci] = c2;
      float hold = (float)hs[m][hidg];
      hnxt[ci] = (_Float16)(on ? h2 : hold);
    }
  }
}

// after 64 steps, final state in pp=0: out = h_fwd + h_bwd (rows split qu/qv)
__global__ void combine_h(const _Float16* __restrict__ hbase,
                          float* __restrict__ qu, float* __restrict__ qv) {
  size_t i = (size_t)blockIdx.x * blockDim.x + threadIdx.x;  // 8192*256
  const _Float16* hf = hbase;                                // dir0, pp0
  const _Float16* hb = hbase + (size_t)2 * 8192 * 256;       // dir1, pp0
  float v = (float)hf[i] + (float)hb[i];
  size_t row = i >> 8;
  if (row < 2048) qu[i] = v;
  else            qv[i - (size_t)2048 * 256] = v;
}

// ---------------------------------------------------------------------------
// NE loss partial reduction (deterministic two-pass, no atomics)
// ---------------------------------------------------------------------------
__global__ void ne_partial(const float* __restrict__ ru, const float* __restrict__ rv,
                           const float* __restrict__ au, const float* __restrict__ av,
                           const int* __restrict__ rpos_u, const int* __restrict__ rpos_v,
                           const int* __restrict__ rpos_neg,
                           const int* __restrict__ apos_u, const int* __restrict__ apos_v,
                           const int* __restrict__ apos_neg,
                           const float* __restrict__ qu, const float* __restrict__ qv,
                           float* __restrict__ p1, float* __restrict__ p2) {
  __shared__ float r1[256], r2[256];
  const int b = blockIdx.x;
  const int d = threadIdx.x;
  float u = ru[(size_t)rpos_u[b] * 256 + d] + au[(size_t)apos_u[b] * 256 + d] +
            qu[(size_t)b * 256 + d];
  float v = rv[(size_t)rpos_v[b] * 256 + d] + av[(size_t)apos_v[b] * 256 + d] +
            qv[(size_t)b * 256 + d];
  float s1 = u * v, s2 = 0.0f;
#pragma unroll
  for (int k = 0; k < 5; ++k) {
    int n = b * 5 + k;
    float nv = rv[(size_t)rpos_neg[n] * 256 + d] + av[(size_t)apos_neg[n] * 256 + d] +
               qv[(size_t)(1024 + n) * 256 + d];
    s2 += nv * u;
  }
  r1[d] = s1; r2[d] = s2;
  __syncthreads();
  for (int off = 128; off; off >>= 1) {
    if (d < off) { r1[d] += r1[d + off]; r2[d] += r2[d + off]; }
    __syncthreads();
  }
  if (d == 0) { p1[b] = r1[0]; p2[b] = r2[0]; }
}

__global__ void ne_final(const float* __restrict__ p1, const float* __restrict__ p2,
                         float* __restrict__ sums) {
  __shared__ float r1[256], r2[256];
  const int t = threadIdx.x;
  float a = 0.0f, b = 0.0f;
#pragma unroll
  for (int j = 0; j < 4; ++j) { a += p1[t + 256 * j]; b += p2[t + 256 * j]; }
  r1[t] = a; r2[t] = b;
  __syncthreads();
  for (int off = 128; off; off >>= 1) {
    if (t < off) { r1[t] += r1[t + off]; r2[t] += r2[t + off]; }
    __syncthreads();
  }
  if (t == 0) { sums[0] = r1[0]; sums[1] = r2[0]; }
}

// ---------------------------------------------------------------------------
// Ranking part: convs reduce to dot products.  One block per row r;
// 256 threads = 32 channels x 8 segments of 32 dims.
// ---------------------------------------------------------------------------
__global__ void rank_kernel(const float* __restrict__ ru, const float* __restrict__ au,
                            const int* __restrict__ rank_r, const int* __restrict__ rank_a,
                            const int* __restrict__ rank_acc,
                            const float* __restrict__ qu,   // rows 1024.. hold eq
                            const float* __restrict__ k1, const float* __restrict__ cb1,
                            const float* __restrict__ k2, const float* __restrict__ cb2,
                            const float* __restrict__ k3, const float* __restrict__ cb3,
                            const float* __restrict__ f1w, const float* __restrict__ f1b,
                            const float* __restrict__ f2w, const float* __restrict__ f2b,
                            const float* __restrict__ f3w, const float* __restrict__ f3b,
                            const float* __restrict__ sums,
                            float* __restrict__ outp) {
  __shared__ float vecs[4][256];       // er, eq, ea, eacc
  __shared__ float part[13][32][8];
  __shared__ float chanL[32], chanH[32];
  const int r = blockIdx.x;
  const int tid = threadIdx.x;

  {
    int ir = rank_r[r], ia = rank_a[r], ic = rank_acc[r];
    vecs[0][tid] = ru[(size_t)ir * 256 + tid];
    vecs[1][tid] = qu[(size_t)(1024 + r) * 256 + tid];
    vecs[2][tid] = au[(size_t)ia * 256 + tid];
    vecs[3][tid] = au[(size_t)ic * 256 + tid];
  }
  __syncthreads();

  const int c = tid & 31, seg = tid >> 5, d0 = seg * 32;
  float p[13];
#pragma unroll
  for (int q = 0; q < 13; ++q) p[q] = 0.0f;
  for (int j = 0; j < 32; ++j) {
    int d = d0 + j;
    float e0 = vecs[0][d], e1 = vecs[1][d], e2 = vecs[2][d], e3 = vecs[3][d];
    float w1  = k1[c * 256 + d];
    p[0] += e0 * w1; p[1] += e1 * w1; p[2] += e2 * w1; p[3] += e3 * w1;
    float w2a = k2[(c * 2 + 0) * 256 + d];
    float w2b = k2[(c * 2 + 1) * 256 + d];
    p[4] += e0 * w2a; p[5] += e1 * w2a;
    p[6] += e1 * w2b; p[7] += e2 * w2b; p[8] += e3 * w2b;
    float w30 = k3[(c * 3 + 0) * 256 + d];
    float w31 = k3[(c * 3 + 1) * 256 + d];
    float w32 = k3[(c * 3 + 2) * 256 + d];
    p[9] += e0 * w30; p[10] += e1 * w31; p[11] += e2 * w32; p[12] += e3 * w32;
  }
#pragma unroll
  for (int q = 0; q < 13; ++q) part[q][c][seg] = p[q];
  __syncthreads();

  if (tid < 32) {
    float t[13];
#pragma unroll
    for (int q = 0; q < 13; ++q) {
      float s = 0.0f;
      for (int g = 0; g < 8; ++g) s += part[q][tid][g];
      t[q] = s;
    }
    float b1 = cb1[tid], b2 = cb2[tid], b3 = cb3[tid];
    float c10 = relu(t[0] + b1), c11 = relu(t[1] + b1);
    float c12 = relu(t[2] + b1), c13 = relu(t[3] + b1);
    float p1l = fmaxf(fmaxf(c10, c11), c12);
    float p1h = fmaxf(fmaxf(c10, c11), c13);
    float l20 = relu(t[4] + t[6] + b2);            // rows (er,eq)
    float l21 = relu(t[5] + t[7] + b2);            // rows (eq,ea)
    float h21 = relu(t[5] + t[8] + b2);            // rows (eq,eacc)
    float p2l = fmaxf(l20, l21), p2h = fmaxf(l20, h21);
    float c3l = relu(t[9] + t[10] + t[11] + b3);
    float c3h = relu(t[9] + t[10] + t[12] + b3);
    chanL[tid] = p1l * f1w[tid] + p2l * f2w[tid] + c3l * f3w[tid];
    // NOTE: reference applies fc3 to convnet1 output on the high branch.
    chanH[tid] = p1h * f3w[tid] + p2h * f2w[tid] + c3h * f3w[tid];
  }
  __syncthreads();

  if (tid == 0) {
    float lo = 0.0f, hs = 0.0f;
    for (int q = 0; q < 32; ++q) { lo += chanL[q]; hs += chanH[q]; }
    lo += f1b[0] + f2b[0] + f3b[0];
    hs += f3b[0] + f2b[0] + f3b[0];
    float ne = logsig(sums[0]) + logsig(-sums[1]);
    outp[r] = ne + logsig(lo - hs);
  }
}

// ---------------------------------------------------------------------------
// launch
// ---------------------------------------------------------------------------
extern "C" void kernel_launch(void* const* d_in, const int* in_sizes, int n_in,
                              void* d_out, int out_size, void* d_ws, size_t ws_size,
                              hipStream_t stream) {
  const float* q_texts  = (const float*)d_in[0];
  const int*   q_lens   = (const int*)d_in[1];
  const int*   rpos_u   = (const int*)d_in[2];
  const int*   rpos_v   = (const int*)d_in[3];
  const int*   rpos_neg = (const int*)d_in[4];
  const int*   apos_u   = (const int*)d_in[5];
  const int*   apos_v   = (const int*)d_in[6];
  const int*   apos_neg = (const int*)d_in[7];
  const int*   qpos_u   = (const int*)d_in[8];
  const int*   qpos_v   = (const int*)d_in[9];
  const int*   qpos_neg = (const int*)d_in[10];
  const int*   rank_r   = (const int*)d_in[11];
  const int*   rank_a   = (const int*)d_in[12];
  const int*   rank_acc = (const int*)d_in[13];
  const int*   rank_q   = (const int*)d_in[14];
  // d_in[15] = nsample (K=5, fixed)
  const float* ru_emb = (const float*)d_in[16];
  const float* rv_emb = (const float*)d_in[17];
  const float* au_emb = (const float*)d_in[18];
  const float* av_emb = (const float*)d_in[19];
  const float* uWih_f = (const float*)d_in[20];
  const float* uWhh_f = (const float*)d_in[21];
  const float* ub_f   = (const float*)d_in[22];
  const float* uWih_b = (const float*)d_in[23];
  const float* uWhh_b = (const float*)d_in[24];
  const float* ub_b   = (const float*)d_in[25];
  const float* vWih_f = (const float*)d_in[26];
  const float* vWhh_f = (const float*)d_in[27];
  const float* vb_f   = (const float*)d_in[28];
  const float* vWih_b = (const float*)d_in[29];
  const float* vWhh_b = (const float*)d_in[30];
  const float* vb_b   = (const float*)d_in[31];
  const float* k1  = (const float*)d_in[32];
  const float* cb1 = (const float*)d_in[33];
  const float* k2  = (const float*)d_in[34];
  const float* cb2 = (const float*)d_in[35];
  const float* k3  = (const float*)d_in[36];
  const float* cb3 = (const float*)d_in[37];
  const float* f1w = (const float*)d_in[38];
  const float* f1b = (const float*)d_in[39];
  const float* f2w = (const float*)d_in[40];
  const float* f2b = (const float*)d_in[41];
  const float* f3w = (const float*)d_in[42];
  const float* f3b = (const float*)d_in[43];
  float* outp = (float*)d_out;

  // workspace layout (~44 MiB)
  const int WN = 1024 * 256;                 // elements per weight matrix
  char* w = (char*)d_ws;
  _Float16* w16 = (_Float16*)w;              // 8 matrices f16 (4 MiB)
  size_t off = (size_t)8 * WN * sizeof(_Float16);
  _Float16* hstate = (_Float16*)(w + off);   // [2][2][8192][256] f16 (16 MiB)
  off += (size_t)2 * 2 * 8192 * 256 * sizeof(_Float16);
  float* cstate = (float*)(w + off);         // [2][8192][256] f32 (16 MiB)
  off += (size_t)2 * 8192 * 256 * sizeof(float);
  float* qu   = (float*)(w + off); off += (size_t)2048 * 256 * sizeof(float);
  float* qv   = (float*)(w + off); off += (size_t)6144 * 256 * sizeof(float);
  int* seqidx = (int*)(w + off);   off += 8192 * sizeof(int);
  int* lens   = (int*)(w + off);   off += 8192 * sizeof(int);
  float* p1   = (float*)(w + off); off += 1024 * sizeof(float);
  float* p2   = (float*)(w + off); off += 1024 * sizeof(float);
  float* sums = (float*)(w + off); off += 2 * sizeof(float);

  // 1) convert all LSTM weights to f16  (order: u ihF,hhF,ihB,hhB; v same)
  const float* wsrc[8] = {uWih_f, uWhh_f, uWih_b, uWhh_b,
                          vWih_f, vWhh_f, vWih_b, vWhh_b};
  for (int i = 0; i < 8; ++i)
    cvt_f32_f16<<<WN / 256, 256, 0, stream>>>(wsrc[i], w16 + (size_t)i * WN, WN);

  // 2) row tables + zero recurrent state
  build_idx<<<32, 256, 0, stream>>>(qpos_u, rank_q, qpos_v, qpos_neg, q_lens,
                                    seqidx, lens);
  zero_state<<<(2 * 2 * 8192 * 256) / 256, 256, 0, stream>>>(hstate, cstate);

  // 3) 64 timesteps; each launch advances all 4 recurrences by one step
  for (int step = 0; step < 64; ++step)
    lstm_step<<<1024, 128, 0, stream>>>(q_texts, seqidx, lens, step, w16,
                                        ub_f, ub_b, vb_f, vb_b, hstate, cstate);

  // 4) out = h_fwd + h_bwd -> qu (rows 0..2047) / qv (rows 2048..8191)
  combine_h<<<(8192 * 256) / 256, 256, 0, stream>>>(hstate, qu, qv);

  // 5) NE loss (deterministic two-pass reduction)
  ne_partial<<<1024, 256, 0, stream>>>(ru_emb, rv_emb, au_emb, av_emb,
      rpos_u, rpos_v, rpos_neg, apos_u, apos_v, apos_neg, qu, qv, p1, p2);
  ne_final<<<1, 256, 0, stream>>>(p1, p2, sums);

  // 6) ranking part + final combine
  rank_kernel<<<1024, 256, 0, stream>>>(ru_emb, au_emb, rank_r, rank_a, rank_acc,
      qu, k1, cb1, k2, cb2, k3, cb3, f1w, f1b, f2w, f2b, f3w, f3b, sums, outp);
}
